// DeepKernelNetwork_65747359367887
// MI455X (gfx1250) — matrix operand order repaired and checked by reading, not compiled, not run on hardware
//
#include <hip/hip_runtime.h>
#include <hip/hip_bf16.h>

typedef __attribute__((ext_vector_type(16))) _Float16     v16h;
typedef __attribute__((ext_vector_type(8)))  float        v8f;
typedef __attribute__((ext_vector_type(4)))  unsigned int u32x4;

#define BATCH 3
#define HH 256
#define WW 256
#define CC 16
#define C2 64                 // C*NB
#define MTOT (BATCH*HH*WW)    // 196608 output pixels per conv
#define EPS 1e-3f

// ---- CDNA5 async copy helpers ---------------------------------------------
// GLOBAL_LOAD_ASYNC_TO_LDS_B128: per-lane 16B global->LDS, tracked by ASYNCcnt.
__device__ __forceinline__ void async_copy_b128(unsigned lds_off, const void* gptr) {
    asm volatile("global_load_async_to_lds_b128 %0, %1, off"
                 :: "v"(lds_off), "v"(gptr) : "memory");
}
__device__ __forceinline__ void wait_asynccnt0() {
    asm volatile("s_wait_asynccnt 0" ::: "memory");
}
__device__ __forceinline__ unsigned lds_off_of(const void* p) {
    // LDS aperture: hardware uses addr[31:0] as the LDS byte address.
    return (unsigned)(unsigned long long)p;
}

// ---------------------------------------------------------------------------
// conv1: 3x3, 16 -> 64, fused bias + BN + ReLU, output fp16
// Implicit GEMM: M = MTOT, K = 144 (padded to 160), N = 64.
// Block = 256 threads = 8 waves; block tile M=128, N=64; wave tile 16x64.
// A needs f32->f16 conversion, so it stays on the VALU gather path.
// ---------------------------------------------------------------------------
__global__ __launch_bounds__(256) void dkn_conv1(
    const float* __restrict__ mr, const float* __restrict__ W1,
    const float* __restrict__ b1, const float* __restrict__ g1,
    const float* __restrict__ be1, const float* __restrict__ mu1,
    const float* __restrict__ v1, _Float16* __restrict__ x1)
{
    __shared__ _Float16 lA[128][40];   // 80B rows: every 16B segment aligned
    __shared__ _Float16 lB[64][36];

    const int tid  = threadIdx.x;
    const int wid  = tid >> 5;
    const int lane = tid & 31;
    const int m0   = blockIdx.x * 128;

    v8f acc[4] = {};

    for (int k0 = 0; k0 < 160; k0 += 32) {
        // ---- stage A chunk (im2col gather, f32 -> f16) ----
        for (int e = tid; e < 128 * 32; e += 256) {
            int r  = e >> 5;
            int kk = e & 31;
            int k  = k0 + kk;
            float val = 0.f;
            if (k < 144) {
                int ci  = k & 15;
                int kxy = k >> 4;
                int kx  = kxy % 3, ky = kxy / 3;
                int m   = m0 + r;
                int b   = m >> 16;
                int rem = m & 65535;
                int y   = rem >> 8, x = rem & 255;
                int yy  = y + ky - 1, xx = x + kx - 1;
                if ((unsigned)yy < 256u && (unsigned)xx < 256u)
                    val = mr[((((b << 8) + yy) << 8) | xx) * CC + ci];
            }
            lA[r][kk] = (_Float16)val;
        }
        // ---- stage B chunk (weights, transposed to [n][k]) ----
        for (int e = tid; e < 64 * 32; e += 256) {
            int n  = e >> 5;
            int kk = e & 31;
            int k  = k0 + kk;
            float w = (k < 144) ? W1[k * C2 + n] : 0.f;
            lB[n][kk] = (_Float16)w;
            if (k + 32 < 144) __builtin_prefetch(&W1[(k + 32) * C2 + n]);  // global_prefetch_b8
        }
        __syncthreads();

        v16h a;
        {
            int m  = (wid << 4) + (lane & 15);
            int kb = (lane >> 4) << 3;
            #pragma unroll
            for (int t = 0; t < 8; ++t) {
                a[t]     = lA[m][kb + t];
                a[8 + t] = lA[m][kb + 16 + t];
            }
        }
        #pragma unroll
        for (int nt = 0; nt < 4; ++nt) {
            v16h bf;
            int n  = (nt << 4) + (lane & 15);
            int kb = (lane >> 4) << 3;
            #pragma unroll
            for (int t = 0; t < 16; ++t) bf[t] = lB[n][kb + t + ((t >> 3) << 3)];
            acc[nt] = __builtin_amdgcn_wmma_f32_16x16x32_f16(
                false, a, false, bf, (short)0, acc[nt], false, false);
        }
        __syncthreads();
    }

    #pragma unroll
    for (int nt = 0; nt < 4; ++nt) {
        int n = (nt << 4) + (lane & 15);
        float s  = g1[n] * rsqrtf(v1[n] + EPS);
        float sh = (b1[n] - mu1[n]) * s + be1[n];
        #pragma unroll
        for (int r = 0; r < 8; ++r) {
            int mrow = ((lane >> 4) << 3) + r;
            int m = m0 + (wid << 4) + mrow;
            float v = acc[nt][r] * s + sh;
            v = v > 0.f ? v : 0.f;
            x1[(size_t)m * C2 + n] = (_Float16)v;
        }
    }
}

// ---------------------------------------------------------------------------
// conv2: 3x3, 64 -> 64, fused bias + BN + group-of-4 softmax + mask, fp16 out
// K = 576 = 18 * 32. A-tile is fp16 raw data -> staged with
// GLOBAL_LOAD_ASYNC_TO_LDS_B128 (ASYNCcnt), zero-fill fallback on borders.
// Each 32-wide K chunk stays within one (ky,kx) tap: ci = (k0 & 32) + kk,
// so each A row chunk is 64 contiguous bytes of x1 -> 4 async b128 per row.
// ---------------------------------------------------------------------------
__global__ __launch_bounds__(256) void dkn_conv2(
    const _Float16* __restrict__ x1, const float* __restrict__ W2,
    const float* __restrict__ b2, const float* __restrict__ g2,
    const float* __restrict__ be2, const float* __restrict__ mu2,
    const float* __restrict__ v2, _Float16* __restrict__ bfeat)
{
    __shared__ _Float16 lA[128][40];   // 80B rows; segments 16B-aligned
    __shared__ _Float16 lB[64][36];

    const int tid  = threadIdx.x;
    const int wid  = tid >> 5;
    const int lane = tid & 31;
    const int m0   = blockIdx.x * 128;

    v8f acc[4] = {};

    for (int k0 = 0; k0 < 576; k0 += 32) {
        const int ci0 = k0 & 32;       // 0 or 32 within the 64-wide tap
        const int kxy = k0 >> 6;       // 0..8
        const int kx  = kxy % 3, ky = kxy / 3;

        // ---- stage A chunk: 128 rows x 64B via async b128 copies ----
        for (int e = tid; e < 512; e += 256) {       // 2 per thread
            int r   = e >> 2;
            int seg = e & 3;                         // 16B segment
            int m   = m0 + r;
            int b   = m >> 16;
            int rem = m & 65535;
            int y   = rem >> 8, x = rem & 255;
            int yy  = y + ky - 1, xx = x + kx - 1;
            unsigned dst = lds_off_of(&lA[r][seg * 8]);
            if ((unsigned)yy < 256u && (unsigned)xx < 256u) {
                const _Float16* g =
                    &x1[(size_t)((((b << 8) + yy) << 8) | xx) * C2 + ci0 + seg * 8];
                async_copy_b128(dst, g);
            } else {
                *(u32x4*)&lA[r][seg * 8] = (u32x4){0u, 0u, 0u, 0u};  // ds_store_b128
            }
        }
        // ---- stage B chunk (weights f32 -> f16, transposed) ----
        for (int e = tid; e < 64 * 32; e += 256) {
            int n  = e >> 5;
            int kk = e & 31;
            int k  = k0 + kk;
            lB[n][kk] = (_Float16)W2[k * C2 + n];
            if (k + 32 < 576) __builtin_prefetch(&W2[(k + 32) * C2 + n]);
        }
        wait_asynccnt0();              // ASYNCcnt -> 0 before the barrier
        __syncthreads();

        v16h a;
        {
            int m  = (wid << 4) + (lane & 15);
            int kb = (lane >> 4) << 3;
            #pragma unroll
            for (int t = 0; t < 8; ++t) {
                a[t]     = lA[m][kb + t];
                a[8 + t] = lA[m][kb + 16 + t];
            }
        }
        #pragma unroll
        for (int nt = 0; nt < 4; ++nt) {
            v16h bf;
            int n  = (nt << 4) + (lane & 15);
            int kb = (lane >> 4) << 3;
            #pragma unroll
            for (int t = 0; t < 16; ++t) bf[t] = lB[n][kb + t + ((t >> 3) << 3)];
            acc[nt] = __builtin_amdgcn_wmma_f32_16x16x32_f16(
                false, a, false, bf, (short)0, acc[nt], false, false);
        }
        __syncthreads();
    }

    // ---- epilogue: BN + width-4 lane-quad softmax + mask, store fp16 ----
    #pragma unroll
    for (int nt = 0; nt < 4; ++nt) {
        int n = (nt << 4) + (lane & 15);
        float s  = g2[n] * rsqrtf(v2[n] + EPS);
        float sh = (b2[n] - mu2[n]) * s + be2[n];
        #pragma unroll
        for (int r = 0; r < 8; ++r) {
            int mrow = ((lane >> 4) << 3) + r;
            int m = m0 + (wid << 4) + mrow;
            float v = acc[nt][r] * s + sh;
            float mx = fmaxf(v, __shfl_xor(v, 1, 4));
            mx = fmaxf(mx, __shfl_xor(mx, 2, 4));
            float e  = __expf(v - mx);
            float sum = e + __shfl_xor(e, 1, 4);
            sum = sum + __shfl_xor(sum, 2, 4);
            float sm = e / sum;
            sm = sm > 1e-5f ? sm : 0.f;
            bfeat[(size_t)m * C2 + n] = (_Float16)sm;
        }
    }
}

// ---------------------------------------------------------------------------
__global__ __launch_bounds__(256) void dkn_zero(float* __restrict__ out, int n)
{
    int i = blockIdx.x * 256 + threadIdx.x;
    if (i < n) out[i] = 0.f;
}

// ---------------------------------------------------------------------------
// patch kernel: per 8x8 patch (stride 4, SAME pad 2). Patch staging is a raw
// copy -> async b128 global->LDS; zero-fill borders with ds_store_b128.
// ---------------------------------------------------------------------------
__global__ __launch_bounds__(256) void dkn_patch(
    const _Float16* __restrict__ bfeat, const float* __restrict__ pet,
    float* __restrict__ out)
{
    __shared__ _Float16 sbp[64][64];   // [pixel][c*4+f], 128B rows
    __shared__ float    spp[64][16];   // [pixel][c], 64B rows
    __shared__ float    sdn[4][64];
    __shared__ float    skp[4][64];

    const int blk = blockIdx.x;        // b*4096 + i*64 + j
    const int b = blk >> 12;
    const int i = (blk >> 6) & 63;
    const int j = blk & 63;
    const int tid = threadIdx.x;

    // stage b_features patch: 64 pixels x 8 async b128 segments
    for (int e = tid; e < 512; e += 256) {
        int pix = e >> 3, seg = e & 7;
        int p = pix >> 3, q = pix & 7;
        int y = i * 4 - 2 + p, x = j * 4 - 2 + q;
        unsigned dst = lds_off_of(&sbp[pix][seg * 8]);
        if ((unsigned)y < 256u && (unsigned)x < 256u) {
            const _Float16* g =
                &bfeat[(size_t)((((b << 8) + y) << 8) | x) * C2 + seg * 8];
            async_copy_b128(dst, g);
        } else {
            *(u32x4*)&sbp[pix][seg * 8] = (u32x4){0u, 0u, 0u, 0u};
        }
    }
    // stage pet patch: 64 pixels x 4 async b128 segments
    for (int e = tid; e < 256; e += 256) {
        int pix = e >> 2, seg = e & 3;
        int p = pix >> 3, q = pix & 7;
        int y = i * 4 - 2 + p, x = j * 4 - 2 + q;
        unsigned dst = lds_off_of(&spp[pix][seg * 4]);
        if ((unsigned)y < 256u && (unsigned)x < 256u) {
            const float* g = &pet[((((b << 8) + y) << 8) | x) * CC + seg * 4];
            async_copy_b128(dst, g);
        } else {
            *(u32x4*)&spp[pix][seg * 4] = (u32x4){0u, 0u, 0u, 0u};
        }
    }
    wait_asynccnt0();
    __syncthreads();

    const int cf = tid & 63;
    const int c  = cf >> 2;
    const int quarter = tid >> 6;

    float dn = 0.f, kp = 0.f;
    #pragma unroll
    for (int t = 0; t < 16; ++t) {
        int pix = quarter * 16 + t;
        float bp = (float)sbp[pix][cf];
        dn += bp;
        kp += spp[pix][c] * bp;
    }
    sdn[quarter][cf] = dn;
    skp[quarter][cf] = kp;
    __syncthreads();

    float DN = sdn[0][cf] + sdn[1][cf] + sdn[2][cf] + sdn[3][cf];
    float KP = skp[0][cf] + skp[1][cf] + skp[2][cf] + skp[3][cf];
    float ratio = (DN == 0.f) ? 0.f : (KP / DN);

    const int ry = (i & 1) ? 2 : -2;   // roll for this stride phase (axis H)
    const int rx = (j & 1) ? 2 : -2;   // roll for this stride phase (axis W)

    #pragma unroll
    for (int t = 0; t < 16; ++t) {
        int pix = quarter * 16 + t;
        int p = pix >> 3, q = pix & 7;
        int y = i * 4 - 2 + p, x = j * 4 - 2 + q;
        if ((unsigned)y < 256u && (unsigned)x < 256u) {
            float kn = ratio * (float)sbp[pix][cf] * 0.25f;
            int yo = (y + ry + 256) & 255;
            int xo = (x + rx + 256) & 255;
            atomicAdd(&out[((((b << 8) + yo) << 8) | xo) * CC + c], kn);
        }
    }
}

// ---------------------------------------------------------------------------
extern "C" void kernel_launch(void* const* d_in, const int* in_sizes, int n_in,
                              void* d_out, int out_size, void* d_ws, size_t ws_size,
                              hipStream_t stream) {
    const float* mr  = (const float*)d_in[0];
    const float* pet = (const float*)d_in[1];
    const float* W1  = (const float*)d_in[2];
    const float* b1  = (const float*)d_in[3];
    const float* g1  = (const float*)d_in[4];
    const float* be1 = (const float*)d_in[5];
    const float* mu1 = (const float*)d_in[6];
    const float* v1  = (const float*)d_in[7];
    const float* W2  = (const float*)d_in[8];
    const float* b2  = (const float*)d_in[9];
    const float* g2  = (const float*)d_in[10];
    const float* be2 = (const float*)d_in[11];
    const float* mu2 = (const float*)d_in[12];
    const float* v2  = (const float*)d_in[13];

    _Float16* x1    = (_Float16*)d_ws;                       // M*64 fp16
    _Float16* bfeat = x1 + (size_t)MTOT * C2;                // M*64 fp16
    float*    out   = (float*)d_out;                         // B*H*W*C f32

    const int mblocks = MTOT / 128;                          // 1536

    dkn_conv1<<<mblocks, 256, 0, stream>>>(mr, W1, b1, g1, be1, mu1, v1, x1);
    dkn_conv2<<<mblocks, 256, 0, stream>>>(x1, W2, b2, g2, be2, mu2, v2, bfeat);

    const int nout = BATCH * HH * WW * CC;                   // 3,145,728
    dkn_zero<<<(nout + 255) / 256, 256, 0, stream>>>(out, nout);

    dkn_patch<<<BATCH * 64 * 64, 256, 0, stream>>>(bfeat, pet, out);
}
